// Spectrogram_59828894433323
// MI455X (gfx1250) — matrix operand order repaired and checked by reading, not compiled, not run on hardware
//
#include <hip/hip_runtime.h>
#include <hip/hip_bf16.h>
#include <math.h>

typedef __attribute__((ext_vector_type(16))) _Float16 v16h;
typedef __attribute__((ext_vector_type(8)))  _Float16 v8h;
typedef __attribute__((ext_vector_type(4)))  _Float16 v4h;
typedef __attribute__((ext_vector_type(8)))  float    v8f;
typedef __attribute__((ext_vector_type(4)))  unsigned int u32x4;
typedef __attribute__((ext_vector_type(8)))  int          i32x8;
typedef __attribute__((ext_vector_type(4)))  int          i32x4;

#define B_BATCH   256
#define SIG_LEN   178
#define NFRAMES   43
#define NFREQ     33
#define SPEC_N    (NFREQ * NFRAMES)     // 1419
#define IMG_HW    178
#define IMG_N     (IMG_HW * IMG_HW)     // 31684
#define C_OUT     64
#define OUT_W     89
#define OUT_POS   (OUT_W * OUT_W)       // 7921
#define POS_PAD   7936                  // 496 * 16
#define POS_TILES 496
#define TILES_PER_BLOCK 8               // 496 = 62 * 8
#define N_CLASSES 5
#define G_CHUNK   32                    // images per chunk
#define N_CHUNKS  (B_BATCH / G_CHUNK)

// ---------------------------------------------------------------- utilities
__global__ void zero_acc_kernel(float* __restrict__ acc) {
    int i = blockIdx.x * blockDim.x + threadIdx.x;
    if (i < B_BATCH * C_OUT) acc[i] = 0.0f;
}

// Fold conv weights over the (replicated) 3 input channels, pad K 49->64, f16.
__global__ void build_weff_kernel(const float* __restrict__ conv_w,
                                  _Float16* __restrict__ weff) {
    int i = blockIdx.x * blockDim.x + threadIdx.x;   // over 64*64
    if (i >= C_OUT * 64) return;
    int ch = i >> 6, kk = i & 63;
    float v = 0.0f;
    if (kk < 49) {
        const float* w = conv_w + ch * 3 * 49;
        v = w[kk] + w[49 + kk] + w[98 + kk];
    }
    weff[i] = (_Float16)v;
}

// ------------------------------------------------------------- spectrogram
// Tukey(8, 0.25) periodic == [0,1,1,1,1,1,1,1]; detrend then window; rfft(64).
__global__ void spec_kernel(const float* __restrict__ x,
                            float* __restrict__ spec) {
    const int b = blockIdx.x;
    const float* sig = x + (size_t)b * SIG_LEN;
    const float scale = sqrtf(1.0f / (178.0f * 7.0f));
    const float w0 = 6.283185307179586f / 64.0f;   // 2*pi/NFFT
    for (int idx = threadIdx.x; idx < SPEC_N; idx += blockDim.x) {
        const int k = idx / NFRAMES;
        const int t = idx - k * NFRAMES;
        float s[8];
        float mean = 0.0f;
        #pragma unroll
        for (int n = 0; n < 8; ++n) { s[n] = sig[t * 4 + n]; mean += s[n]; }
        mean *= 0.125f;
        float re = 0.0f, im = 0.0f;
        #pragma unroll
        for (int n = 1; n < 8; ++n) {              // window zeroes n==0
            const float f = s[n] - mean;
            const float ang = w0 * (float)(k * n);
            re += f * cosf(ang);
            im -= f * sinf(ang);
        }
        spec[(size_t)b * SPEC_N + idx] = sqrtf(re * re + im * im) * scale;
    }
}

// ------------------------------------------- bilinear resize 33x43 -> 178x178
__global__ void resize_kernel(const float* __restrict__ spec,
                              float* __restrict__ img) {
    const int b = blockIdx.x;
    const float* sp = spec + (size_t)b * SPEC_N;
    float* im = img + (size_t)b * IMG_N;
    const float sy_scale = (float)NFREQ  / (float)IMG_HW;
    const float sx_scale = (float)NFRAMES / (float)IMG_HW;
    for (int idx = threadIdx.x; idx < IMG_N; idx += blockDim.x) {
        const int y = idx / IMG_HW, xo = idx - y * IMG_HW;
        const float sy = (y + 0.5f) * sy_scale - 0.5f;
        const float sx = (xo + 0.5f) * sx_scale - 0.5f;
        const float fy = floorf(sy), fx = floorf(sx);
        const float wy = sy - fy,    wx = sx - fx;
        int y0 = (int)fy, x0 = (int)fx;
        int y0c = y0 < 0 ? 0 : (y0 > NFREQ - 1 ? NFREQ - 1 : y0);
        int y1 = y0 + 1;
        int y1c = y1 < 0 ? 0 : (y1 > NFREQ - 1 ? NFREQ - 1 : y1);
        int x0c = x0 < 0 ? 0 : (x0 > NFRAMES - 1 ? NFRAMES - 1 : x0);
        int x1 = x0 + 1;
        int x1c = x1 < 0 ? 0 : (x1 > NFRAMES - 1 ? NFRAMES - 1 : x1);
        const float v00 = sp[y0c * NFRAMES + x0c];
        const float v01 = sp[y0c * NFRAMES + x1c];
        const float v10 = sp[y1c * NFRAMES + x0c];
        const float v11 = sp[y1c * NFRAMES + x1c];
        im[idx] = (1.0f - wy) * ((1.0f - wx) * v00 + wx * v01)
                +          wy * ((1.0f - wx) * v10 + wx * v11);
    }
}

// ------------------------------------------------------------------ im2col
// col[bl][pos][K] f16, K padded 49->64 (zeros), pos padded 7921->7936 (zeros).
// K-contiguous layout: lanes take k = 16*hh + e for B, the same order A uses.
__global__ void im2col_kernel(const float* __restrict__ img,
                              _Float16* __restrict__ col, int bG) {
    const int bl = blockIdx.y;
    const float* im = img + (size_t)(bG + bl) * IMG_N;
    const int idx = blockIdx.x * blockDim.x + threadIdx.x;  // < 7936*16
    const int pos = idx >> 4;
    const int kg  = idx & 15;           // group of 4 K values
    const bool valid = pos < OUT_POS;
    const int oy  = pos / OUT_W;
    const int ox  = pos - oy * OUT_W;
    const int iy0 = oy * 2 - 3;
    const int ix0 = ox * 2 - 3;
    v4h out;
    #pragma unroll
    for (int t = 0; t < 4; ++t) {
        const int K = kg * 4 + t;
        float v = 0.0f;
        if (valid && K < 49) {
            const int kh = K / 7, kw = K - kh * 7;
            const int iy = iy0 + kh, ix = ix0 + kw;
            if (iy >= 0 && iy < IMG_HW && ix >= 0 && ix < IMG_HW)
                v = im[iy * IMG_HW + ix];
        }
        out[t] = (_Float16)v;
    }
    *(v4h*)(col + ((size_t)bl * POS_PAD + pos) * 64 + kg * 4) = out;
}

// ---------------------------------------- GEMM conv + bias + ReLU + mean-sum
// D(16ch x 16pos) = A(16ch x 64) * B(64 x 16pos), two v_wmma_f32_16x16x32_f16.
// Weights broadcast to LDS via the Tensor Data Mover (async-tensor engine).
__global__ __launch_bounds__(128)
void conv_wmma_kernel(const _Float16* __restrict__ weff,
                      const _Float16* __restrict__ col,
                      const float* __restrict__ conv_b,
                      float* __restrict__ acc, int bG) {
    __shared__ _Float16 lds_w[C_OUT * 64];       // only LDS object -> offset 0

    const int bl   = blockIdx.y;
    const int b    = bG + bl;
    const int lane = threadIdx.x & 31;
    const int wave = threadIdx.x >> 5;           // 0..3 -> channel tile
    const int n    = lane & 15;                  // column within tile
    const int hh   = lane >> 4;                  // lane half
    const int ch_base = wave * 16;

#if __has_builtin(__builtin_amdgcn_tensor_load_to_lds) && \
    __has_builtin(__builtin_amdgcn_s_wait_tensorcnt)
    if (wave == 0) {                             // one wave issues the TDM op
        const unsigned long long ga = (unsigned long long)(uintptr_t)weff;
        u32x4 g0;
        g0[0] = 1u;                              // count=1 valid descriptor
        g0[1] = 0u;                              // lds_addr = 0
        g0[2] = (unsigned int)(ga & 0xFFFFFFFFu);
        g0[3] = (unsigned int)((ga >> 32) & 0x01FFFFFFu) | (2u << 30); // type=2
        i32x8 g1;
        g1[0] = 0x10000;      // data_size = 1 (2-byte elements)
        g1[1] = 0x00400000;   // tensor_dim0 = 64 (low16 in [31:16])
        g1[2] = 0x00400000;   // tensor_dim0 hi = 0, tensor_dim1 = 64
        g1[3] = 0x00400000;   // tensor_dim1 hi = 0, tile_dim0 = 64
        g1[4] = 64;           // tile_dim1 = 64, tile_dim2 = 0
        g1[5] = 64;           // tensor_dim0_stride = 64 (low 32)
        g1[6] = 0x10000000;   // stride0 hi = 0, tensor_dim1_stride lo16 = 4096
        g1[7] = 0;            // stride1 hi
        i32x4 gz4 = {0, 0, 0, 0};
        i32x8 gz8 = {0, 0, 0, 0, 0, 0, 0, 0};
        __builtin_amdgcn_tensor_load_to_lds(g0, g1, gz4, gz4, gz8, 0);
        __builtin_amdgcn_s_wait_tensorcnt(0);
    }
    __syncthreads();
#else
    for (int i = threadIdx.x; i < C_OUT * 64 / 8; i += blockDim.x)
        ((v8h*)lds_w)[i] = ((const v8h*)weff)[i];
    __syncthreads();
#endif

    // ---- A from LDS: lane needs K = hh*16+e (a0) and 32+hh*16+e (a1) --
    const int ch = ch_base + n;
    const v8h* wp = (const v8h*)lds_w;
    const v8h a0lo = wp[ch * 8 + hh * 2];
    const v8h a0hi = wp[ch * 8 + hh * 2 + 1];
    const v8h a1lo = wp[ch * 8 + hh * 2 + 4];
    const v8h a1hi = wp[ch * 8 + hh * 2 + 5];
    const v16h a0 = __builtin_shufflevector(a0lo, a0hi, 0,1,2,3,4,5,6,7,
                                            8,9,10,11,12,13,14,15);
    const v16h a1 = __builtin_shufflevector(a1lo, a1hi, 0,1,2,3,4,5,6,7,
                                            8,9,10,11,12,13,14,15);

    float bias[8];
    #pragma unroll
    for (int r = 0; r < 8; ++r) bias[r] = conv_b[ch_base + r + 8 * hh];

    float sum[8];
    #pragma unroll
    for (int r = 0; r < 8; ++r) sum[r] = 0.0f;

    #pragma unroll 2
    for (int jt = 0; jt < TILES_PER_BLOCK; ++jt) {
        const int tile = blockIdx.x * TILES_PER_BLOCK + jt;
        const int pos  = tile * 16 + n;
        const bool valid = pos < OUT_POS;

        // ---- B from col: lane needs K = hh*16+e (b0) and 32+hh*16+e (b1) --
        const v8h* bp = (const v8h*)(col + ((size_t)bl * POS_PAD + pos) * 64);
        const v8h b0lo = bp[hh * 2];
        const v8h b0hi = bp[hh * 2 + 1];
        const v8h b1lo = bp[hh * 2 + 4];
        const v8h b1hi = bp[hh * 2 + 5];
        const v16h b0 = __builtin_shufflevector(b0lo, b0hi, 0,1,2,3,4,5,6,7,
                                                8,9,10,11,12,13,14,15);
        const v16h b1 = __builtin_shufflevector(b1lo, b1hi, 0,1,2,3,4,5,6,7,
                                                8,9,10,11,12,13,14,15);

        v8f c = {};
        c = __builtin_amdgcn_wmma_f32_16x16x32_f16(false, a0, false, b0,
                                                   (short)0, c, false, false);
        c = __builtin_amdgcn_wmma_f32_16x16x32_f16(false, a1, false, b1,
                                                   (short)0, c, false, false);

        #pragma unroll
        for (int r = 0; r < 8; ++r) {
            float v = c[r] + bias[r];
            v = v > 0.0f ? v : 0.0f;
            sum[r] += valid ? v : 0.0f;
        }
    }

    // reduce the 16 position columns within each lane-half, then atomics
    #pragma unroll
    for (int r = 0; r < 8; ++r) {
        float s = sum[r];
        s += __shfl_xor(s, 1, 32);
        s += __shfl_xor(s, 2, 32);
        s += __shfl_xor(s, 4, 32);
        s += __shfl_xor(s, 8, 32);
        if (n == 0)
            atomicAdd(&acc[b * C_OUT + ch_base + r + 8 * hh], s);
    }
}

// ---------------------------------------------------------------- mean + FC
__global__ void fc_kernel(const float* __restrict__ acc,
                          const float* __restrict__ fc_w,
                          const float* __restrict__ fc_b,
                          float* __restrict__ out) {
    __shared__ float a[C_OUT];
    const int b = blockIdx.x;
    const int t = threadIdx.x;
    a[t] = acc[b * C_OUT + t] * (1.0f / (float)OUT_POS);
    __syncthreads();
    if (t < N_CLASSES) {
        float s = fc_b[t];
        #pragma unroll
        for (int ch = 0; ch < C_OUT; ++ch)
            s += a[ch] * fc_w[t * C_OUT + ch];
        out[b * N_CLASSES + t] = s;
    }
}

// ---------------------------------------------------------------- launcher
extern "C" void kernel_launch(void* const* d_in, const int* in_sizes, int n_in,
                              void* d_out, int out_size, void* d_ws, size_t ws_size,
                              hipStream_t stream) {
    const float* x      = (const float*)d_in[0];  // (256,1,178)
    const float* conv_w = (const float*)d_in[1];  // (64,3,7,7)
    const float* conv_b = (const float*)d_in[2];  // (64,)
    const float* fc_w   = (const float*)d_in[3];  // (5,64)
    const float* fc_b   = (const float*)d_in[4];  // (5,)
    float* out = (float*)d_out;                   // (256,5)

    char* ws = (char*)d_ws;
    size_t off = 0;
    float*    acc  = (float*)(ws + off);  off += (size_t)B_BATCH * C_OUT * 4;
    _Float16* weff = (_Float16*)(ws + off); off += (size_t)C_OUT * 64 * 2;
    float*    spec = (float*)(ws + off);  off += (size_t)B_BATCH * SPEC_N * 4;
    float*    img  = (float*)(ws + off);  off += (size_t)B_BATCH * IMG_N * 4;
    _Float16* col  = (_Float16*)(ws + off); // G_CHUNK * 7936 * 64 f16 (~31 MB)

    zero_acc_kernel  <<<dim3((B_BATCH * C_OUT + 255) / 256), 256, 0, stream>>>(acc);
    build_weff_kernel<<<dim3((C_OUT * 64 + 255) / 256), 256, 0, stream>>>(conv_w, weff);
    spec_kernel      <<<dim3(B_BATCH), 256, 0, stream>>>(x, spec);
    resize_kernel    <<<dim3(B_BATCH), 256, 0, stream>>>(spec, img);

    for (int c = 0; c < N_CHUNKS; ++c) {
        const int bG = c * G_CHUNK;
        im2col_kernel   <<<dim3(POS_PAD * 16 / 256, G_CHUNK), 256, 0, stream>>>
                         (img, col, bG);
        conv_wmma_kernel<<<dim3(POS_TILES / TILES_PER_BLOCK, G_CHUNK), 128, 0, stream>>>
                         (weff, col, conv_b, acc, bG);
    }

    fc_kernel<<<dim3(B_BATCH), C_OUT, 0, stream>>>(acc, fc_w, fc_b, out);
}
